// MSVA_auto_17377437680004
// MI455X (gfx1250) — matrix operand order repaired and checked by reading, not compiled, not run on hardware
//
#include <hip/hip_runtime.h>
#include <hip/hip_bf16.h>

typedef __attribute__((ext_vector_type(16))) __bf16 v16bf;
typedef __attribute__((ext_vector_type(8)))  float  v8f;

#define NSEQ  4096
#define MDIM  1024
#define APER  250
#define BANDW 512   // padded stride for band storage (499 valid columns)

// ---- CDNA5 async global->LDS DMA path (guarded; sync fallback otherwise) ----
#if defined(__has_builtin)
#  if __has_builtin(__builtin_amdgcn_global_load_async_to_lds_b128)
#    define GEMM_ASYNC 1
#  endif
#endif
#ifndef GEMM_ASYNC
#  define GEMM_ASYNC 0
#endif

#if GEMM_ASYNC
typedef int async_v4i __attribute__((vector_size(16)));             // matches builtin param
typedef __attribute__((address_space(1))) async_v4i* g_v4i_ptr;     // global int4*
typedef __attribute__((address_space(3))) async_v4i* l_v4i_ptr;     // LDS int4*
#endif

static __device__ __forceinline__ void stage_b128(const __bf16* __restrict__ g,
                                                  __bf16* l) {
#if GEMM_ASYNC
  __builtin_amdgcn_global_load_async_to_lds_b128((g_v4i_ptr)g, (l_v4i_ptr)l, 0, 0);
#else
  *(uint4*)l = *(const uint4*)g;
#endif
}

static __device__ __forceinline__ void wait_stage() {
#if GEMM_ASYNC
#  if __has_builtin(__builtin_amdgcn_s_wait_asynccnt)
  __builtin_amdgcn_s_wait_asynccnt(0);
#  else
  asm volatile("s_wait_asynccnt 0x0" ::: "memory");
#  endif
#endif
}

// ---------------- block reductions ----------------
static __device__ __forceinline__ float blk_reduce_sum(float v, float* sm) {
  int tid = threadIdx.x;
  sm[tid] = v; __syncthreads();
  for (int s = blockDim.x >> 1; s > 0; s >>= 1) {
    if (tid < s) sm[tid] += sm[tid + s];
    __syncthreads();
  }
  float r = sm[0]; __syncthreads();
  return r;
}
static __device__ __forceinline__ float blk_reduce_max(float v, float* sm) {
  int tid = threadIdx.x;
  sm[tid] = v; __syncthreads();
  for (int s = blockDim.x >> 1; s > 0; s >>= 1) {
    if (tid < s) sm[tid] = fmaxf(sm[tid], sm[tid + s]);
    __syncthreads();
  }
  float r = sm[0]; __syncthreads();
  return r;
}

// ---------------- fp32 -> bf16 conversion ----------------
__global__ void k_f32_to_bf16(const float* __restrict__ in, __bf16* __restrict__ out, int n) {
  for (int i = blockIdx.x * blockDim.x + threadIdx.x; i < n; i += gridDim.x * blockDim.x)
    out[i] = (__bf16)in[i];
}

// ---------------- generic bf16 WMMA GEMM:  C[M,N] = A[M,K] @ W[N,K]^T ----------------
// 8 waves per block, 128x64 C tile, double-buffered LDS panels (A:128x32, B:64x32),
// async global->LDS DMA (ASYNCcnt) when available, k-step = 32.
__global__ void k_gemm_wmma(const __bf16* __restrict__ A, const __bf16* __restrict__ W,
                            const float* __restrict__ bias,
                            float* __restrict__ Cf, __bf16* __restrict__ Cbf,
                            int Md, int Nd, int Kd, float scale, int relu, int out_bf) {
  __shared__ __align__(16) __bf16 As[2][128 * 32];
  __shared__ __align__(16) __bf16 Bs[2][64 * 32];

  const int tid  = threadIdx.x;
  const int wave = tid >> 5;
  const int lane = tid & 31;
  const int l    = lane & 15;
  const int h    = lane >> 4;
  const int m0   = blockIdx.x * 128;
  const int n0g  = blockIdx.y * 64;
  const int nk   = Kd >> 5;

  v8f acc[4];
#pragma unroll
  for (int t = 0; t < 4; ++t)
#pragma unroll
    for (int e = 0; e < 8; ++e) acc[t][e] = 0.0f;

  // ---- stage k-panel kt into LDS buffer `buf` (3 x b128 per thread) ----
#define STAGE_PANEL(kt, buf)                                                     \
  do {                                                                           \
    const int k0s = (kt) * 32;                                                   \
    _Pragma("unroll")                                                            \
    for (int idx = tid; idx < 768; idx += 256) {                                 \
      if (idx < 512) {                                                           \
        const int rr = idx >> 2, seg = idx & 3;                                  \
        stage_b128(&A[(size_t)(m0 + rr) * Kd + k0s + seg * 8],                   \
                   &As[buf][rr * 32 + seg * 8]);                                 \
      } else {                                                                   \
        const int rr = (idx - 512) >> 2, seg = idx & 3;                          \
        stage_b128(&W[(size_t)(n0g + rr) * Kd + k0s + seg * 8],                  \
                   &Bs[buf][rr * 32 + seg * 8]);                                 \
      }                                                                          \
    }                                                                            \
  } while (0)

  STAGE_PANEL(0, 0);

  for (int kt = 0; kt < nk; ++kt) {
    const int b = kt & 1;
    wait_stage();        // own async DMAs for buffer b complete
    __syncthreads();     // everyone's DMAs/stores visible; previous buffer free

    if (kt + 1 < nk) STAGE_PANEL(kt + 1, b ^ 1);

    // A fragment: lane half h holds K = {h*8..h*8+7, 16+h*8..16+h*8+7} of row (r0+l)
    union { v16bf v; uint4 q[2]; } af, bfrag;
    const int r0 = wave * 16;
    af.q[0] = *(const uint4*)&As[b][(r0 + l) * 32 + h * 8];
    af.q[1] = *(const uint4*)&As[b][(r0 + l) * 32 + 16 + h * 8];

#pragma unroll
    for (int ct = 0; ct < 4; ++ct) {
      // B fragment: lane half h holds K = {h*8..h*8+7, 16+h*8..16+h*8+7} of weight row (ct*16+l)
      bfrag.q[0] = *(const uint4*)&Bs[b][(ct * 16 + l) * 32 + h * 8];
      bfrag.q[1] = *(const uint4*)&Bs[b][(ct * 16 + l) * 32 + 16 + h * 8];
      acc[ct] = __builtin_amdgcn_wmma_f32_16x16x32_bf16(
          false, af.v, false, bfrag.v, (short)0, acc[ct], false, false);
    }
  }
#undef STAGE_PANEL

  // epilogue: C layout -> VGPR r, lane: (M = r + 8h, N = l)
#pragma unroll
  for (int ct = 0; ct < 4; ++ct) {
#pragma unroll
    for (int r = 0; r < 8; ++r) {
      const int gm = m0 + wave * 16 + r + 8 * h;
      const int gn = n0g + ct * 16 + l;
      float v = acc[ct][r] * scale + (bias ? bias[gn] : 0.0f);
      if (relu) v = fmaxf(v, 0.0f);
      if (out_bf) Cbf[(size_t)gm * Nd + gn] = (__bf16)v;
      else        Cf [(size_t)gm * Nd + gn] = v;
    }
  }
}

// ---------------- banded Q@K^T via WMMA (one wave per 16x16 tile) ----------------
__global__ void k_band_qk_wmma(const __bf16* __restrict__ Q, const __bf16* __restrict__ Kb,
                               float* __restrict__ S, int n) {
  const int i0 = blockIdx.y * 16;
  const int j0 = i0 + ((int)blockIdx.x - 16) * 16;
  if (j0 < 0 || j0 >= n) return;
  if (j0 > i0 + 15 + (APER - 1) || j0 + 15 < i0 - (APER - 1)) return;

  const int lane = threadIdx.x & 31;
  const int l = lane & 15, h = lane >> 4;

  v8f acc;
#pragma unroll
  for (int e = 0; e < 8; ++e) acc[e] = 0.0f;

  for (int k0 = 0; k0 < MDIM; k0 += 32) {
    union { v16bf v; uint4 q[2]; } af, bfrag;
    af.q[0]    = *(const uint4*)&Q [(size_t)(i0 + l) * MDIM + k0 + h * 8];
    af.q[1]    = *(const uint4*)&Q [(size_t)(i0 + l) * MDIM + k0 + 16 + h * 8];
    bfrag.q[0] = *(const uint4*)&Kb[(size_t)(j0 + l) * MDIM + k0 + h * 8];
    bfrag.q[1] = *(const uint4*)&Kb[(size_t)(j0 + l) * MDIM + k0 + 16 + h * 8];
    acc = __builtin_amdgcn_wmma_f32_16x16x32_bf16(
        false, af.v, false, bfrag.v, (short)0, acc, false, false);
  }

#pragma unroll
  for (int r = 0; r < 8; ++r) {
    const int i = i0 + r + 8 * h;
    const int j = j0 + l;
    const int d = j - i;
    if (d > -APER && d < APER)
      S[(size_t)i * BANDW + d + (APER - 1)] = acc[r];
  }
}

// ---------------- row softmax over the band ----------------
__global__ void k_band_softmax(float* __restrict__ S, int n) {
  __shared__ float sm[256];
  const int i = blockIdx.x, t = threadIdx.x;
  const int jlo = (i - (APER - 1) < 0) ? 0 : i - (APER - 1);
  const int jhi = (i + (APER - 1) > n - 1) ? n - 1 : i + (APER - 1);
  const int clo = jlo - i + (APER - 1);
  const int chi = jhi - i + (APER - 1);

  float m = -3.4e38f;
  for (int c = clo + t; c <= chi; c += 256) m = fmaxf(m, S[(size_t)i * BANDW + c]);
  m = blk_reduce_max(m, sm);

  float s = 0.0f;
  for (int c = clo + t; c <= chi; c += 256) {
    float e = expf(S[(size_t)i * BANDW + c] - m);
    S[(size_t)i * BANDW + c] = e;
    s += e;
  }
  s = blk_reduce_sum(s, sm);
  const float inv = 1.0f / s;
  for (int c = clo + t; c <= chi; c += 256) S[(size_t)i * BANDW + c] *= inv;
}

// ---------------- scatter band -> dense att1 (with explicit zeros) ----------------
__global__ void k_write_att(const float* __restrict__ S, float* __restrict__ att) {
  const int i = blockIdx.x;
  for (int j = threadIdx.x; j < NSEQ; j += blockDim.x) {
    const int d = j - i;
    float v = 0.0f;
    if (d > -APER && d < APER) v = S[(size_t)i * BANDW + d + (APER - 1)];
    att[(size_t)i * NSEQ + j] = v;
  }
}

// ---------------- y = att.T @ V (banded), bf16 output ----------------
__global__ void k_attT_v(const float* __restrict__ S, const float* __restrict__ V,
                         __bf16* __restrict__ T, int n) {
  const int i = blockIdx.y;
  const int m = blockIdx.x * 256 + threadIdx.x;
  const int jlo = (i - (APER - 1) < 0) ? 0 : i - (APER - 1);
  const int jhi = (i + (APER - 1) > n - 1) ? n - 1 : i + (APER - 1);
  float s = 0.0f;
  for (int j = jlo; j <= jhi; ++j)
    s = fmaf(S[(size_t)j * BANDW + (i - j + (APER - 1))], V[(size_t)j * MDIM + m], s);
  T[(size_t)i * MDIM + m] = (__bf16)s;
}

// ---------------- residual + LayerNorm(1024, ddof=1), accumulate into y_sum ----------------
__global__ void k_ln_res(const float* __restrict__ Y, const float* __restrict__ X,
                         const float* __restrict__ g, const float* __restrict__ b,
                         float* __restrict__ out, int accumulate) {
  __shared__ float sm[256];
  const int i = blockIdx.x, t = threadIdx.x;
  float v[4], s = 0.0f;
#pragma unroll
  for (int k = 0; k < 4; ++k) {
    const int c = t + k * 256;
    v[k] = Y[(size_t)i * MDIM + c] + X[(size_t)i * MDIM + c];
    s += v[k];
  }
  const float mean = blk_reduce_sum(s, sm) * (1.0f / 1024.0f);
  float sq = 0.0f;
#pragma unroll
  for (int k = 0; k < 4; ++k) { const float d = v[k] - mean; sq += d * d; }
  const float var = blk_reduce_sum(sq, sm) * (1.0f / 1023.0f);   // unbiased
  const float inv = 1.0f / (sqrtf(var) + 1e-6f);
#pragma unroll
  for (int k = 0; k < 4; ++k) {
    const int c = t + k * 256;
    const float r = g[c] * (v[k] - mean) * inv + b[c];
    out[(size_t)i * MDIM + c] = accumulate ? (out[(size_t)i * MDIM + c] + r) : r;
  }
}

// ---------------- final LN(128, ddof=1) + Linear(128->1) + sigmoid ----------------
__global__ void k_final(const float* __restrict__ H, const float* __restrict__ g,
                        const float* __restrict__ b, const float* __restrict__ kdw,
                        const float* __restrict__ kdb, float* __restrict__ out) {
  __shared__ float sm[128];
  const int i = blockIdx.x, t = threadIdx.x;
  const float v = H[(size_t)i * 128 + t];
  const float mean = blk_reduce_sum(v, sm) * (1.0f / 128.0f);
  const float d = v - mean;
  const float var = blk_reduce_sum(d * d, sm) * (1.0f / 127.0f);
  const float inv = 1.0f / (sqrtf(var) + 1e-6f);
  const float ln = g[t] * d * inv + b[t];
  const float dot = blk_reduce_sum(ln * kdw[t], sm);
  if (t == 0) out[i] = 1.0f / (1.0f + expf(-(dot + kdb[0])));
}

// ---------------- host orchestration ----------------
extern "C" void kernel_launch(void* const* d_in, const int* in_sizes, int n_in,
                              void* d_out, int out_size, void* d_ws, size_t ws_size,
                              hipStream_t stream) {
  (void)in_sizes; (void)n_in; (void)out_size; (void)ws_size;

  const float* x[3] = { (const float*)d_in[0], (const float*)d_in[1], (const float*)d_in[2] };
  const float* Wq   = (const float*)d_in[3];
  const float* Wk   = (const float*)d_in[4];
  const float* Wv   = (const float*)d_in[5];
  const float* Wo   = (const float*)d_in[6];
  const float* ka_w = (const float*)d_in[7];
  const float* ka_b = (const float*)d_in[8];
  const float* kb_w = (const float*)d_in[9];
  const float* kb_b = (const float*)d_in[10];
  const float* kc_w = (const float*)d_in[11];
  const float* kc_b = (const float*)d_in[12];
  const float* kd_w = (const float*)d_in[13];
  const float* kd_b = (const float*)d_in[14];
  const float* g13  = (const float*)d_in[15];
  const float* b13  = (const float*)d_in[16];
  const float* gkc  = (const float*)d_in[17];
  const float* bkc  = (const float*)d_in[18];

  float* out_pred = (float*)d_out;            // [4096]
  float* out_att  = (float*)d_out + NSEQ;     // [4096 x 4096]

  // ---- workspace layout ----
  char* ws = (char*)d_ws;
  constexpr size_t SZ_WBF  = (size_t)MDIM * MDIM * 2;        // 2 MB (bf16 1024x1024)
  constexpr size_t SZ_KB   = (size_t)512 * 1024 * 2;         // 1 MB
  constexpr size_t SZ_KC   = (size_t)128 * 512 * 2;          // 128 KB
  constexpr size_t SZ_ACT4 = (size_t)NSEQ * MDIM * 4;        // 16 MB fp32 activations
  constexpr size_t SZ_ACT2 = (size_t)NSEQ * MDIM * 2;        // 8 MB bf16 activations
  constexpr size_t SZ_BAND = (size_t)NSEQ * BANDW * 4;       // 8 MB band buffer

  size_t o = 0;
  auto take = [&](size_t n) { size_t r = o; o += (n + 255) & ~(size_t)255; return r; };
  const size_t o_wq  = take(SZ_WBF), o_wk = take(SZ_WBF), o_wv = take(SZ_WBF), o_wo = take(SZ_WBF);
  const size_t o_ka  = take(SZ_WBF), o_kb = take(SZ_KB),  o_kc = take(SZ_KC);
  const size_t o_ysum= take(SZ_ACT4);
  const size_t o_xbf = take(SZ_ACT2);
  const size_t o_qbf = take(SZ_ACT2);
  const size_t o_kbf = take(SZ_ACT2);
  const size_t o_vf  = take(SZ_ACT4);
  const size_t o_S   = take(SZ_BAND);
  const size_t o_tbf = take(SZ_ACT2);
  const size_t o_ystr= take(SZ_ACT4);

  __bf16* wq_bf = (__bf16*)(ws + o_wq);
  __bf16* wk_bf = (__bf16*)(ws + o_wk);
  __bf16* wv_bf = (__bf16*)(ws + o_wv);
  __bf16* wo_bf = (__bf16*)(ws + o_wo);
  __bf16* ka_bf = (__bf16*)(ws + o_ka);
  __bf16* kb_bf = (__bf16*)(ws + o_kb);
  __bf16* kc_bf = (__bf16*)(ws + o_kc);
  float*  y_sum = (float* )(ws + o_ysum);
  __bf16* x_bf  = (__bf16*)(ws + o_xbf);
  __bf16* q_bf  = (__bf16*)(ws + o_qbf);
  __bf16* k_bf  = (__bf16*)(ws + o_kbf);
  float*  v_f   = (float* )(ws + o_vf);
  float*  Sband = (float* )(ws + o_S);
  __bf16* t_bf  = (__bf16*)(ws + o_tbf);
  float*  y_str = (float* )(ws + o_ystr);
  // MLP phase reuses per-stream scratch:
  __bf16* ysum_bf = x_bf;                 // 8 MB
  __bf16* h1_bf   = q_bf;                 // 4096x1024 bf16, 8 MB
  __bf16* h2_bf   = k_bf;                 // 4096x512 bf16, 4 MB
  float*  h3_f    = v_f;                  // 4096x128 fp32, 2 MB

  auto cvt = [&](const float* in, __bf16* outp, int n) {
    k_f32_to_bf16<<<2048, 256, 0, stream>>>(in, outp, n);
  };
  auto gemm = [&](const __bf16* A, const __bf16* W, const float* bias,
                  float* Cf, __bf16* Cbf, int Md, int Nd, int Kd,
                  float scale, int relu, int obf) {
    dim3 grid(Md / 128, Nd / 64);
    k_gemm_wmma<<<grid, 256, 0, stream>>>(A, W, bias, Cf, Cbf, Md, Nd, Kd, scale, relu, obf);
  };

  // weights -> bf16
  cvt(Wq, wq_bf, MDIM * MDIM);
  cvt(Wk, wk_bf, MDIM * MDIM);
  cvt(Wv, wv_bf, MDIM * MDIM);
  cvt(Wo, wo_bf, MDIM * MDIM);
  cvt(ka_w, ka_bf, 1024 * 1024);
  cvt(kb_w, kb_bf, 512 * 1024);
  cvt(kc_w, kc_bf, 128 * 512);

  for (int s = 0; s < 3; ++s) {
    cvt(x[s], x_bf, NSEQ * MDIM);
    gemm(x_bf, wk_bf, nullptr, nullptr, k_bf, NSEQ, MDIM, MDIM, 1.0f,  0, 1);
    gemm(x_bf, wq_bf, nullptr, nullptr, q_bf, NSEQ, MDIM, MDIM, 0.06f, 0, 1);
    gemm(x_bf, wv_bf, nullptr, v_f,     nullptr, NSEQ, MDIM, MDIM, 1.0f, 0, 0);

    k_band_qk_wmma<<<dim3(33, NSEQ / 16), 32, 0, stream>>>(q_bf, k_bf, Sband, NSEQ);
    k_band_softmax<<<NSEQ, 256, 0, stream>>>(Sband, NSEQ);
    if (s == 0) k_write_att<<<NSEQ, 256, 0, stream>>>(Sband, out_att);
    k_attT_v<<<dim3(MDIM / 256, NSEQ), 256, 0, stream>>>(Sband, v_f, t_bf, NSEQ);

    gemm(t_bf, wo_bf, nullptr, y_str, nullptr, NSEQ, MDIM, MDIM, 1.0f, 0, 0);
    k_ln_res<<<NSEQ, 256, 0, stream>>>(y_str, x[s], g13, b13, y_sum, s > 0);
  }

  // MLP head
  cvt(y_sum, ysum_bf, NSEQ * MDIM);
  gemm(ysum_bf, ka_bf, ka_b, nullptr, h1_bf, NSEQ, 1024, 1024, 1.0f, 0, 1);
  gemm(h1_bf,   kb_bf, kb_b, nullptr, h2_bf, NSEQ, 512,  1024, 1.0f, 0, 1);
  gemm(h2_bf,   kc_bf, kc_b, h3_f,   nullptr, NSEQ, 128, 512,  1.0f, 1, 0);
  k_final<<<NSEQ, 128, 0, stream>>>(h3_f, gkc, bkc, kd_w, kd_b, out_pred);
}
